// ENRR_40303973105912
// MI455X (gfx1250) — matrix operand order repaired and checked by reading, not compiled, not run on hardware
//
#include <hip/hip_runtime.h>
#include <hip/hip_bf16.h>
#include <math.h>
#include <stdint.h>

// ---------------------------------------------------------------------------
// CDNA5 (gfx1250) implementation of the event-cloud scoring network.
// GEMMs: v_wmma_f32_16x16x32_f16, one wave per 16x16 C tile, 4 waves/block.
//   - B (weights) pre-packed into per-lane v16h fragments (aligned b128 loads)
//   - A loads branch-free (index clamp + zero select), full/tail K split
// KNN fused (LDS candidate tiles + register top-K insertion sort).
// Selective scan: one wave per (batch, channel), shfl_xor reduction over state.
// ---------------------------------------------------------------------------

#define BB   2
#define NP0  4096
#define NP1  2048
#define KL   16
#define KSP  9

typedef _Float16 v16h __attribute__((ext_vector_type(16)));
typedef float    v8f  __attribute__((ext_vector_type(8)));

__device__ __forceinline__ float dev_silu(float x) { return x / (1.f + __expf(-x)); }

__device__ __forceinline__ float act_apply(float x, int act) {
  if (act == 1) return x > 0.f ? x : 0.f;                  // relu
  if (act == 2) return (x > 20.f) ? x : log1pf(__expf(x)); // softplus
  return x;
}

// ---------------- weight pack: W[K,N] f32 -> per-lane B fragments (f16) -----
// layout: frag[((kt*NT + nt)*32 + lane)*16 + e], e-th half per ISA B 32x16 map
__global__ void pack_b_kernel(const float* __restrict__ W, int ldw, int K, int N,
                              int NT, _Float16* __restrict__ out, int total)
{
  int i = blockIdx.x * blockDim.x + threadIdx.x;
  if (i >= total) return;              // total = KT*NT*32
  int lane = i & 31; int t = i >> 5;
  int nt = t % NT;  int kt = t / NT;
  int half = lane >> 4, mn = lane & 15;
  int bn = nt * 16 + mn;
  _Float16* o = out + (size_t)i * 16;
  #pragma unroll
  for (int e = 0; e < 16; ++e) {
    int kk = kt * 32 + ((e < 8) ? (8 * half + e) : (16 + 8 * half + (e - 8)));
    float v = (kk < K && bn < N) ? W[(size_t)kk * ldw + bn] : 0.f;
    o[e] = (_Float16)v;
  }
}

// ---------------- WMMA GEMM:  C[M,N] = act(A[M,K] * W[K,N] + bias) ----------
// REQUIRES: M multiple of 16. Bp = packed B fragments. 4 waves per block.
__global__ void wmma_gemm_kernel(const float* __restrict__ A, int lda,
                                 const _Float16* __restrict__ Bp, int NT,
                                 const float* __restrict__ bias,
                                 float* __restrict__ C, int ldc,
                                 int M, int N, int K, int act)
{
  const int wave = threadIdx.x >> 5;
  const int lane = threadIdx.x & 31;
  const int half = lane >> 4;
  const int mn   = lane & 15;
  const int m0 = (blockIdx.x * 4 + wave) * 16;
  const int nt = blockIdx.y;
  if (m0 >= M) return;                         // wave-uniform exit

  const float* arow = A + (size_t)(m0 + mn) * lda;  // always in range (M % 16 == 0)
  const _Float16* bbase = Bp + ((size_t)nt * 32 + lane) * 16;
  const size_t bstep = (size_t)NT * 32 * 16;

  v8f acc = {};
  const int kfull = K >> 5;
  for (int kt = 0; kt < kfull; ++kt) {          // no bounds logic at all
    const int k0 = kt * 32;
    v16h afrag;
    #pragma unroll
    for (int e = 0; e < 8; ++e)
      afrag[e] = (_Float16)arow[k0 + 8 * half + e];
    #pragma unroll
    for (int e = 0; e < 8; ++e)
      afrag[8 + e] = (_Float16)arow[k0 + 16 + 8 * half + e];
    v16h bfrag = *(const v16h*)(bbase + (size_t)kt * bstep);
    acc = __builtin_amdgcn_wmma_f32_16x16x32_f16(false, afrag, false, bfrag,
                                                 (short)0, acc, false, false);
  }
  if (K & 31) {                                 // tail: clamp index, select zero
    const int k0 = kfull * 32;
    v16h afrag;
    #pragma unroll
    for (int e = 0; e < 16; ++e) {
      int kk = k0 + ((e < 8) ? (8 * half + e) : (16 + 8 * half + (e - 8)));
      int kc = kk < K ? kk : 0;                 // always-valid address
      float v = arow[kc];
      afrag[e] = (_Float16)(kk < K ? v : 0.f);
    }
    v16h bfrag = *(const v16h*)(bbase + (size_t)kfull * bstep);
    acc = __builtin_amdgcn_wmma_f32_16x16x32_f16(false, afrag, false, bfrag,
                                                 (short)0, acc, false, false);
  }

  const int n = nt * 16 + mn;
  if (n < N) {
    const float bv = bias ? bias[n] : 0.f;
    #pragma unroll
    for (int v = 0; v < 8; ++v) {
      int m = m0 + v + 8 * half;
      float x = acc[v] + bv;
      C[(size_t)m * ldc + n] = act_apply(x, act);
    }
  }
}

// ---------------- fused KNN (top-K smallest sqdist, index output) -----------
template<int KMAX>
__global__ void knn_kernel(const float* __restrict__ q, int ldq, int Nq,
                           const float* __restrict__ p, int ldp, int Np,
                           int* __restrict__ outidx, float radius2)
{
  __shared__ float sx[128], sy[128], sz[128];
  const int b = blockIdx.y;
  const int tid = threadIdx.x;
  const int i = blockIdx.x * 128 + tid;
  const float* qb = q + (size_t)b * Nq * ldq;
  const float* pb = p + (size_t)b * Np * ldp;
  const bool active = (i < Nq);
  float qx = 0.f, qy = 0.f, qz = 0.f;
  if (active) { qx = qb[(size_t)i*ldq]; qy = qb[(size_t)i*ldq+1]; qz = qb[(size_t)i*ldq+2]; }

  float bd[KMAX]; int bi[KMAX];
  #pragma unroll
  for (int k = 0; k < KMAX; ++k) { bd[k] = 3.4e38f; bi[k] = 0; }

  for (int j0 = 0; j0 < Np; j0 += 128) {
    int j = j0 + tid;
    if (j < Np) {
      const float* pr = pb + (size_t)j * ldp;
      sx[tid] = pr[0]; sy[tid] = pr[1]; sz[tid] = pr[2];
      if (j + 128 < Np) __builtin_prefetch(pb + (size_t)(j + 128) * ldp, 0, 1);
    }
    __syncthreads();
    int lim = Np - j0; if (lim > 128) lim = 128;
    if (active) {
      for (int jj = 0; jj < lim; ++jj) {
        float dx = qx - sx[jj], dy = qy - sy[jj], dz = qz - sz[jj];
        float d = dx*dx + dy*dy + dz*dz;
        if (d < bd[KMAX-1]) {
          float dd = d; int ji = j0 + jj;
          #pragma unroll
          for (int k = 0; k < KMAX; ++k) {
            if (dd < bd[k]) { float td = bd[k]; int ti = bi[k];
                              bd[k] = dd; bi[k] = ji; dd = td; ji = ti; }
          }
        }
      }
    }
    __syncthreads();
  }
  if (active) {
    #pragma unroll
    for (int k = 0; k < KMAX; ++k) {
      int id = bi[k];
      if (radius2 > 0.f && bd[k] > radius2) id = i;
      outidx[((size_t)b * Nq + i) * KMAX + k] = id;
    }
  }
}

// ---------------- gathers ---------------------------------------------------
__global__ void gather_kernel(const float* __restrict__ src, int lds, int coff, int C,
                              const int* __restrict__ idx, int K, int Np, int Nq,
                              float* __restrict__ out, long long total)
{
  long long i = (long long)blockIdx.x * blockDim.x + threadIdx.x;
  if (i >= total) return;
  int c = (int)(i % C); long long t = i / C;
  int k = (int)(t % K); long long bn = t / K;
  int b = (int)(bn / Nq);
  int id = idx[bn * K + k];
  out[i] = src[((size_t)b * Np + id) * lds + coff + c];
}

__global__ void gather_sub_kernel(const float* __restrict__ src, int lds, int coff, int C,
                                  const int* __restrict__ idx, int K, int Np,
                                  float* __restrict__ out, long long total)
{
  long long i = (long long)blockIdx.x * blockDim.x + threadIdx.x;
  if (i >= total) return;
  int c = (int)(i % C); long long t = i / C;
  int k = (int)(t % K); long long bn = t / K;
  int b = (int)(bn / Np); int n = (int)(bn % Np);
  int id = idx[bn * K + k];
  float qv = src[((size_t)b * Np + n)  * lds + coff + c];
  float pv = src[((size_t)b * Np + id) * lds + coff + c];
  out[i] = qv - pv;
}

// ---------------- layernorm (thread per row) --------------------------------
__global__ void ln_kernel(const float* __restrict__ in, float* __restrict__ out,
                          const float* __restrict__ g, const float* __restrict__ b,
                          long long R, int C)
{
  long long r = (long long)blockIdx.x * blockDim.x + threadIdx.x;
  if (r >= R) return;
  const float* x = in + r * C;
  float m = 0.f;
  for (int c = 0; c < C; ++c) m += x[c];
  m /= (float)C;
  float v = 0.f;
  for (int c = 0; c < C; ++c) { float d = x[c] - m; v += d * d; }
  v /= (float)C;
  float rs = rsqrtf(v + 1e-5f);
  float* o = out + r * C;
  for (int c = 0; c < C; ++c) o[c] = (x[c] - m) * rs * g[c] + b[c];
}

__global__ void combine_kernel(const float* __restrict__ lt, int ldlt,
                               const float* __restrict__ psi,
                               const float* __restrict__ delta,
                               float* __restrict__ out, long long total, int K, int C)
{
  long long i = (long long)blockIdx.x * blockDim.x + threadIdx.x;
  if (i >= total) return;
  int c = (int)(i % C); long long t = i / C; long long bn = t / K;
  out[i] = lt[bn * ldlt + c] - psi[i] + delta[i];
}

__global__ void attn_reduce_kernel(const float* __restrict__ lg,
                                   const float* __restrict__ alpha,
                                   const float* __restrict__ delta,
                                   float* __restrict__ out, long long BN, int C, float scale)
{
  long long i = (long long)blockIdx.x * blockDim.x + threadIdx.x;
  if (i >= BN * C) return;
  int c = (int)(i % C); long long bn = i / C;
  long long base = bn * KL * C + c;
  float v[KL]; float mx = -3.4e38f;
  #pragma unroll
  for (int k = 0; k < KL; ++k) { v[k] = lg[base + (long long)k * C] * scale; mx = fmaxf(mx, v[k]); }
  float s = 0.f;
  #pragma unroll
  for (int k = 0; k < KL; ++k) { v[k] = __expf(v[k] - mx); s += v[k]; }
  float inv = 1.f / s, acc = 0.f;
  #pragma unroll
  for (int k = 0; k < KL; ++k) {
    long long o = base + (long long)k * C;
    acc += (v[k] * inv) * (alpha[o] + delta[o]);
  }
  out[i] = acc;
}

__global__ void dwconv_silu_kernel(const float* __restrict__ xz, int ld,
                                   const float* __restrict__ w, const float* __restrict__ cb,
                                   float* __restrict__ xi, int L, int di, long long total)
{
  long long i = (long long)blockIdx.x * blockDim.x + threadIdx.x;
  if (i >= total) return;
  int c = (int)(i % di); long long t = i / di;
  int l = (int)(t % L); int b = (int)(t / L);
  float acc = cb[c];
  #pragma unroll
  for (int j = 0; j < 4; ++j) {
    int l2 = l - 3 + j;
    if (l2 >= 0) acc += xz[((size_t)b * L + l2) * ld + c] * w[c * 4 + j];
  }
  xi[i] = dev_silu(acc);
}

__global__ void scan_kernel(const float* __restrict__ dt, const float* __restrict__ xi,
                            const float* __restrict__ dbc, int lddbc, int boff, int coff,
                            const float* __restrict__ A_log, const float* __restrict__ Dp,
                            float* __restrict__ y, int L, int di)
{
  int bd = blockIdx.x; int b = bd / di; int d = bd % di;
  int lane = threadIdx.x & 31;
  int s1 = lane, s2 = lane + 32;
  float ea1 = __expf(A_log[(size_t)d * 64 + s1]);
  float ea2 = __expf(A_log[(size_t)d * 64 + s2]);
  float h1 = 0.f, h2 = 0.f;
  float Dv = Dp[d];
  for (int l = 0; l < L; ++l) {
    size_t row = (size_t)b * L + l;
    float dtv = dt[row * di + d];
    float xiv = xi[row * di + d];
    const float* dr = dbc + row * lddbc;
    float a1 = __expf(-dtv * ea1), a2 = __expf(-dtv * ea2);
    h1 = a1 * h1 + dtv * dr[boff + s1] * xiv;
    h2 = a2 * h2 + dtv * dr[boff + s2] * xiv;
    float v = h1 * dr[coff + s1] + h2 * dr[coff + s2];
    #pragma unroll
    for (int m = 16; m >= 1; m >>= 1) v += __shfl_xor(v, m, 32);
    if (lane == 0) y[row * di + d] = v + Dv * xiv;
  }
}

__global__ void mul_silu_kernel(const float* __restrict__ y, const float* __restrict__ xz,
                                int ld, int off, int di, float* __restrict__ out, long long total)
{
  long long i = (long long)blockIdx.x * blockDim.x + threadIdx.x;
  if (i >= total) return;
  int c = (int)(i % di); long long row = i / di;
  float z = xz[(size_t)row * ld + off + c];
  out[i] = y[i] * dev_silu(z);
}

__global__ void add_kernel(const float* a, const float* b, float* o, long long total)
{
  long long i = (long long)blockIdx.x * blockDim.x + threadIdx.x;
  if (i < total) o[i] = a[i] + b[i];
}

__global__ void fte_kernel(const float* __restrict__ xytp,
                           const float* pw, const float* pb,
                           const float* nw, const float* nb_,
                           float* __restrict__ fcat, long long BN, int C)
{
  long long i = (long long)blockIdx.x * blockDim.x + threadIdx.x;
  if (i >= BN * C) return;
  int c = (int)(i % C); long long bn = i / C;
  float ts  = xytp[bn * 4 + 0];
  float pol = xytp[bn * 4 + 3];
  fcat[bn * 64 + 32 + c] = (ts * pw[c] + pb[c]) * pol + (ts * nw[c] + nb_[c]) * (1.f - pol);
}

__global__ void zerot_kernel(const float* __restrict__ xytp, float* __restrict__ c0,
                             long long total)
{
  long long i = (long long)blockIdx.x * blockDim.x + threadIdx.x;
  if (i >= total) return;
  c0[i] = ((i & 3) == 0) ? 0.f : xytp[i];
}

__global__ void slice2_kernel(const float* __restrict__ in, float* __restrict__ out,
                              int Nq, int Np, long long total)
{
  long long i = (long long)blockIdx.x * blockDim.x + threadIdx.x;
  if (i >= total) return;
  int c = (int)(i % 4); long long t = i / 4;
  int n = (int)(t % Nq); int b = (int)(t / Nq);
  out[i] = in[((size_t)b * Np + 2 * n) * 4 + c];
}

__global__ void maxk_kernel(const float* __restrict__ in, float* __restrict__ out,
                            int K, int C, long long total)
{
  long long i = (long long)blockIdx.x * blockDim.x + threadIdx.x;
  if (i >= total) return;
  int c = (int)(i % C); long long bm = i / C;
  float m = -3.4e38f;
  for (int k = 0; k < K; ++k) m = fmaxf(m, in[(bm * K + k) * C + c]);
  out[i] = m;
}

__global__ void softmax1d_kernel(const float* __restrict__ lg, float* __restrict__ aw, int N)
{
  __shared__ float red[256];
  int b = blockIdx.x, tid = threadIdx.x;
  const float* x = lg + (size_t)b * N;
  float mx = -3.4e38f;
  for (int n = tid; n < N; n += 256) mx = fmaxf(mx, x[n]);
  red[tid] = mx; __syncthreads();
  for (int s = 128; s > 0; s >>= 1) { if (tid < s) red[tid] = fmaxf(red[tid], red[tid + s]); __syncthreads(); }
  mx = red[0]; __syncthreads();
  float sm = 0.f;
  for (int n = tid; n < N; n += 256) sm += __expf(x[n] - mx);
  red[tid] = sm; __syncthreads();
  for (int s = 128; s > 0; s >>= 1) { if (tid < s) red[tid] += red[tid + s]; __syncthreads(); }
  float inv = 1.f / red[0];
  for (int n = tid; n < N; n += 256) aw[(size_t)b * N + n] = __expf(x[n] - mx) * inv;
}

__global__ void wsum_kernel(const float* __restrict__ f, const float* __restrict__ aw,
                            float* __restrict__ s, int N, int C)
{
  int b = blockIdx.x, c = threadIdx.x;
  if (c >= C) return;
  float acc = 0.f;
  for (int n = 0; n < N; ++n) acc += f[((size_t)b * N + n) * C + c] * aw[(size_t)b * N + n];
  s[b * C + c] = acc;
}

__global__ void head_kernel(const float* __restrict__ s, const float* __restrict__ w,
                            const float* __restrict__ b0, float* __restrict__ out,
                            int Bn, int C)
{
  int b = threadIdx.x;
  if (b >= Bn) return;
  float a = b0[0];
  for (int c = 0; c < C; ++c) a += s[b * C + c] * w[c];
  out[b] = 1.f / (1.f + __expf(-a));
}

// ---------------------------------------------------------------------------
// Host orchestration
// ---------------------------------------------------------------------------
static inline int cdiv(long long a, long long b) { return (int)((a + b - 1) / b); }

struct BlockParams {
  const float *ln1g, *ln1b, *pe1w, *pe1b, *pe2w, *pe2b;
  const float *ltw, *ltb, *lng, *lnb;
  const float *A_log, *Dp, *conv_b, *conv_w, *dt_b, *dt_w, *in_proj, *out_proj, *x_proj;
  const float *ln2g, *ln2b, *m1w, *m1b, *m2w, *m2b;
  int D, di, r;
};

struct Wsp {
  float *REL, *PH, *DELTA, *PSI, *ALPHA, *LT;
  float *F, *T1, *T2, *T3, *H, *H2;
  float *XZ, *HID, *XI, *DBC, *DT, *YS, *Y2, *FCAT, *C0, *XY2, *LOG, *AW, *S;
  _Float16 *PACK;
  int *IDX;
};

static void gemm(hipStream_t s, Wsp& w, const float* A, int lda,
                 const float* W, int ldw, const float* bias,
                 float* C, int ldc, int M, int N, int K, int act)
{
  int NT = cdiv(N, 16), KT = cdiv(K, 32);
  int ptotal = KT * NT * 32;
  pack_b_kernel<<<cdiv(ptotal, 128), 128, 0, s>>>(W, ldw, K, N, NT, w.PACK, ptotal);
  dim3 g(cdiv(M, 64), NT);
  wmma_gemm_kernel<<<g, 128, 0, s>>>(A, lda, w.PACK, NT, bias, C, ldc, M, N, K, act);
}
#define EW(total) cdiv((total), 256), 256, 0, stream

static void run_block(hipStream_t stream, Wsp& w, const float* coords, int Nn,
                      const BlockParams& P)
{
  const int D = P.D, di = P.di, r = P.r;
  const long long BN  = (long long)BB * Nn;
  const long long BNK = BN * KL;
  const int lddbc = r + 128;

  knn_kernel<KL><<<dim3(cdiv(Nn, 128), BB), 128, 0, stream>>>(coords, 4, Nn, coords, 4, Nn, w.IDX, -1.f);
  ln_kernel<<<EW(BN)>>>(w.F, w.T1, P.ln1g, P.ln1b, BN, D);

  // local transformer on T1
  gather_sub_kernel<<<EW(BNK * 4)>>>(coords, 4, 0, 4, w.IDX, KL, Nn, w.REL, BNK * 4);
  gemm(stream, w, w.REL, 4, P.pe1w, D, P.pe1b, w.PH, D, (int)BNK, D, 4, 1);
  gemm(stream, w, w.PH, D, P.pe2w, D, P.pe2b, w.DELTA, D, (int)BNK, D, D, 0);
  gemm(stream, w, w.T1, D, P.ltw, 3 * D, P.ltb, w.LT, 3 * D, (int)BN, 3 * D, D, 0);
  gather_kernel<<<EW(BNK * D)>>>(w.LT, 3 * D, D,     D, w.IDX, KL, Nn, Nn, w.PSI,   BNK * D);
  gather_kernel<<<EW(BNK * D)>>>(w.LT, 3 * D, 2 * D, D, w.IDX, KL, Nn, Nn, w.ALPHA, BNK * D);
  combine_kernel<<<EW(BNK * D)>>>(w.LT, 3 * D, w.PSI, w.DELTA, w.PH, BNK * D, KL, D);
  ln_kernel<<<EW(BNK)>>>(w.PH, w.PH, P.lng, P.lnb, BNK, D);
  attn_reduce_kernel<<<EW(BN * D)>>>(w.PH, w.ALPHA, w.DELTA, w.H, BN, D, rsqrtf((float)D));

  // mamba(H) + F -> H2
  gemm(stream, w, w.H, D, P.in_proj, 2 * di, nullptr, w.XZ, 2 * di, (int)BN, 2 * di, D, 0);
  dwconv_silu_kernel<<<EW(BN * di)>>>(w.XZ, 2 * di, P.conv_w, P.conv_b, w.XI, Nn, di, BN * di);
  gemm(stream, w, w.XI, di, P.x_proj, lddbc, nullptr, w.DBC, lddbc, (int)BN, lddbc, di, 0);
  gemm(stream, w, w.DBC, lddbc, P.dt_w, di, P.dt_b, w.DT, di, (int)BN, di, r, 2);
  scan_kernel<<<BB * di, 32, 0, stream>>>(w.DT, w.XI, w.DBC, lddbc, r, r + 64, P.A_log, P.Dp, w.YS, Nn, di);
  mul_silu_kernel<<<EW(BN * di)>>>(w.YS, w.XZ, 2 * di, di, di, w.Y2, BN * di);
  gemm(stream, w, w.Y2, di, P.out_proj, D, nullptr, w.T2, D, (int)BN, D, di, 0);
  add_kernel<<<EW(BN * D)>>>(w.T2, w.F, w.H2, BN * D);

  // mlp(ln(H2)) + H2 -> F
  ln_kernel<<<EW(BN)>>>(w.H2, w.T1, P.ln2g, P.ln2b, BN, D);
  gemm(stream, w, w.T1, D, P.m1w, 4 * D, P.m1b, w.HID, 4 * D, (int)BN, 4 * D, D, 1);
  gemm(stream, w, w.HID, 4 * D, P.m2w, D, P.m2b, w.T3, D, (int)BN, D, 4 * D, 0);
  add_kernel<<<EW(BN * D)>>>(w.T3, w.H2, w.F, BN * D);
}

extern "C" void kernel_launch(void* const* d_in, const int* in_sizes, int n_in,
                              void* d_out, int out_size, void* d_ws, size_t ws_size,
                              hipStream_t stream)
{
  (void)in_sizes; (void)n_in; (void)out_size; (void)ws_size;
  #define P(i) ((const float*)d_in[i])
  const float* xytp = P(0);

  // ---- workspace bump allocation (256B aligned blocks) ----
  char* wp = (char*)d_ws;
  auto AB = [&](size_t bytes) -> void* {
    wp = (char*)(((uintptr_t)wp + 255) & ~(uintptr_t)255);
    void* r = (void*)wp; wp += bytes; return r;
  };
  auto AF = [&](size_t n) { return (float*)AB(n * sizeof(float)); };
  Wsp w;
  w.REL   = AF(524288);   w.PH    = AF(4194304); w.DELTA = AF(4194304);
  w.PSI   = AF(4194304);  w.ALPHA = AF(4194304); w.LT    = AF(786432);
  w.F  = AF(524288); w.T1 = AF(524288); w.T2 = AF(524288);
  w.T3 = AF(524288); w.H  = AF(524288); w.H2 = AF(524288);
  w.XZ = AF(1048576); w.HID = AF(1048576); w.XI = AF(524288);
  w.DBC = AF(1064960); w.DT = AF(524288); w.YS = AF(524288); w.Y2 = AF(524288);
  w.FCAT = AF(524288); w.C0 = AF(32768); w.XY2 = AF(16384);
  w.LOG = AF(4096); w.AW = AF(4096); w.S = AF(128);
  w.PACK = (_Float16*)AB(65536 * sizeof(_Float16));
  w.IDX  = (int*)AB(131072 * sizeof(int));

  const long long BN0 = (long long)BB * NP0;

  // ---------------- embed ----------------
  zerot_kernel<<<EW(BN0 * 4)>>>(xytp, w.C0, BN0 * 4);
  const float radius = 5.0f / 260.0f;
  knn_kernel<KSP><<<dim3(cdiv(NP0, 128), BB), 128, 0, stream>>>(
      w.C0, 4, NP0, w.C0, 4, NP0, w.IDX, radius * radius);
  gather_sub_kernel<<<EW(BN0 * KSP * 2)>>>(xytp, 4, 1, 2, w.IDX, KSP, NP0, w.REL, BN0 * KSP * 2);
  gemm(stream, w, w.REL, 18, P(9), 64, P(8), w.HID, 64, (int)BN0, 64, 18, 1);   // sp_mlp.l1 relu
  gemm(stream, w, w.HID, 64, P(11), 32, P(10), w.FCAT, 64, (int)BN0, 32, 64, 0); // sp_mlp.l2 -> Fsp
  fte_kernel<<<EW(BN0 * 32)>>>(xytp, P(6), P(5), P(4), P(3), w.FCAT, BN0, 32);
  gemm(stream, w, w.FCAT, 64, P(7), 32, nullptr, w.F, 32, (int)BN0, 32, 64, 0);  // proj_w
  ln_kernel<<<EW(BN0)>>>(w.F, w.F, P(2), P(1), BN0, 32);                          // embed_norm

  // ---------------- stage 0 block (D=32) ----------------
  BlockParams b0;
  b0.lng = P(21); b0.lnb = P(20); b0.ltb = P(22); b0.ltw = P(23);
  b0.A_log = P(24); b0.Dp = P(25); b0.conv_b = P(26); b0.conv_w = P(27);
  b0.dt_b = P(28); b0.dt_w = P(29); b0.in_proj = P(30); b0.out_proj = P(31); b0.x_proj = P(32);
  b0.m1b = P(33); b0.m1w = P(34); b0.m2b = P(35); b0.m2w = P(36);
  b0.ln1b = P(37); b0.ln1g = P(38); b0.ln2b = P(39); b0.ln2g = P(40);
  b0.pe1b = P(41); b0.pe1w = P(42); b0.pe2b = P(43); b0.pe2w = P(44);
  b0.D = 32; b0.di = 64; b0.r = 2;
  run_block(stream, w, xytp, NP0, b0);

  // ---------------- transition down ----------------
  const long long BN1 = (long long)BB * NP1;
  slice2_kernel<<<EW(BN1 * 4)>>>(xytp, w.XY2, NP1, NP0, BN1 * 4);
  knn_kernel<KL><<<dim3(cdiv(NP1, 128), BB), 128, 0, stream>>>(
      w.XY2, 4, NP1, xytp, 4, NP0, w.IDX, -1.f);
  gather_kernel<<<EW(BN1 * KL * 32)>>>(w.F, 32, 0, 32, w.IDX, KL, NP0, NP1, w.PH, BN1 * KL * 32);
  gemm(stream, w, w.PH, 32, P(46), 64, P(45), w.DELTA, 64, (int)(BN1 * KL), 64, 32, 1);
  maxk_kernel<<<EW(BN1 * 64)>>>(w.DELTA, w.F, KL, 64, BN1 * 64);

  // ---------------- stage 1 block (D=64) ----------------
  BlockParams b1;
  b1.lng = P(48); b1.lnb = P(47); b1.ltb = P(49); b1.ltw = P(50);
  b1.A_log = P(51); b1.Dp = P(52); b1.conv_b = P(53); b1.conv_w = P(54);
  b1.dt_b = P(55); b1.dt_w = P(56); b1.in_proj = P(57); b1.out_proj = P(58); b1.x_proj = P(59);
  b1.m1b = P(60); b1.m1w = P(61); b1.m2b = P(62); b1.m2w = P(63);
  b1.ln1b = P(64); b1.ln1g = P(65); b1.ln2b = P(66); b1.ln2g = P(67);
  b1.pe1b = P(68); b1.pe1w = P(69); b1.pe2b = P(70); b1.pe2w = P(71);
  b1.D = 64; b1.di = 128; b1.r = 4;
  run_block(stream, w, w.XY2, NP1, b1);

  // ---------------- score head ----------------
  gemm(stream, w, w.F,  64, P(15), 64, P(14), w.T1, 64, (int)BN1, 64, 64, 1);
  gemm(stream, w, w.T1, 64, P(17), 64, P(16), w.T2, 64, (int)BN1, 64, 64, 1);
  gemm(stream, w, w.T2, 64, P(13), 1,  P(12), w.LOG, 1, (int)BN1, 1, 64, 0);
  softmax1d_kernel<<<BB, 256, 0, stream>>>(w.LOG, w.AW, NP1);
  wsum_kernel<<<BB, 64, 0, stream>>>(w.T2, w.AW, w.S, NP1, 64);
  head_kernel<<<1, 32, 0, stream>>>(w.S, P(19), P(18), (float*)d_out, BB, 64);
  #undef P
}
